// EfficientGraphAttentionBlock_26938034880917
// MI455X (gfx1250) — matrix operand order repaired and checked by reading, not compiled, not run on hardware
//
#include <hip/hip_runtime.h>

#define NN 10000
#define KK 32
#define HH 128
#define EE 128

typedef __attribute__((ext_vector_type(16))) _Float16 v16h;
typedef __attribute__((ext_vector_type(8)))  float    v8f;
typedef __attribute__((ext_vector_type(4))) unsigned int u32x4;
typedef __attribute__((ext_vector_type(8))) int i32x8;
typedef __attribute__((ext_vector_type(4))) int i32x4;

union AFu { v16h v; uint4 q[2]; };

#if defined(__HIP_DEVICE_COMPILE__) && __has_builtin(__builtin_amdgcn_tensor_load_to_lds) && __has_builtin(__builtin_amdgcn_s_wait_tensorcnt)
#define HAVE_TDM 1
#else
#define HAVE_TDM 0
#endif

// branch-free gelu (tanh form): tanh(u) = sign(u)*(1-e)/(1+e), e = exp(-2|u|)
__device__ __forceinline__ float gelu_f(float x) {
  float u = 0.7978845608028654f * x * (1.0f + 0.044715f * x * x);
  float e = __expf(-2.0f * fabsf(u));
  float t = (1.0f - e) * __builtin_amdgcn_rcpf(1.0f + e);
  t = copysignf(t, u);
  return 0.5f * x * (1.0f + t);
}

// ---- A fragments for 16-row tile, hoisted across all column tiles ----
template<int KT>
struct AFr { v16h a[KT]; };

template<int KT>
__device__ __forceinline__ AFr<KT> load_afrags(const _Float16* A, int lda, int lane) {
  AFr<KT> f;
  int r  = lane & 15;
  int kb = (lane < 16) ? 0 : 8;   // 16-bit A layout: halves hold K {0..7,16..23} / {8..15,24..31}
  const _Float16* ap0 = A + r * lda + kb;
#pragma unroll
  for (int kc = 0; kc < KT; ++kc) {
    AFu t;
    t.q[0] = *(const uint4*)(ap0 + kc * 32);
    t.q[1] = *(const uint4*)(ap0 + kc * 32 + 16);
    f.a[kc] = t.v;
  }
  return f;
}

template<int KT>
__device__ __forceinline__ v8f wmma_btile(const AFr<KT>& af, const _Float16* B, int ldb, int lane) {
  v8f acc = {};
#pragma unroll
  for (int kc = 0; kc < KT; ++kc) {
    AFu fb;
    const _Float16* bp = B + (kc * 32 + ((lane >> 4) << 3)) * ldb + (lane & 15);   // B: lane = N col, K {0..7,16..23} / {8..15,24..31}
#pragma unroll
    for (int s = 0; s < 8; ++s) {
      fb.v[s] = bp[s * ldb];
      fb.v[8 + s] = bp[(16 + s) * ldb];
    }
    acc = __builtin_amdgcn_wmma_f32_16x16x32_f16(false, af.a[kc], false, fb.v,
                                                 (short)0, acc, false, false);
  }
  return acc;
}

// Cooperative f16 weight copy global->LDS (fallback path)
__device__ __forceinline__ void load_w_lds(const _Float16* __restrict__ src, _Float16* dst,
                                           int rows, int cols, int ldd, int tid) {
  int cpr = cols >> 3;
  int chunks = rows * cpr;
  for (int idx = tid; idx < chunks; idx += 256) {
    int r = idx / cpr;
    int c = (idx - r * cpr) << 3;
    *(uint4*)(dst + r * ldd + c) = *(const uint4*)(src + r * cols + c);
  }
}

#if HAVE_TDM
// TDM: DMA a rows x cols f16 weight (contiguous in global) into LDS, optionally
// inserting 4-DWORD row padding (ldw = cols + 8 halves) via D# pad fields.
__device__ __forceinline__ void tdm_load_w(const _Float16* src, _Float16* dst,
                                           int rows, int cols, int ldw) {
  unsigned long long ga = (unsigned long long)(size_t)src;
  unsigned int ldsOff = (unsigned int)(size_t)dst;    // low 32 bits of flat LDS addr = offset
  u32x4 g0;
  g0.x = 1u;                                          // count=1, user descriptor
  g0.y = ldsOff;                                      // lds_addr
  g0.z = (unsigned int)(ga & 0xffffffffu);            // global_addr[31:0]
  g0.w = (unsigned int)((ga >> 32) & 0x1ffffffu) | (2u << 30);  // global_addr[56:32], type=2
  int padEn = (ldw != cols) ? 1 : 0;
  int padInt = (cols == 128) ? 5 : 6;                 // pad every 256B / 512B
  int padAmt = 3;                                     // 4 DWORDs = 8 halves
  i32x8 g1 = {};
  g1[0] = (1 << 16) | (padEn << 20) | ((padEn ? padInt : 0) << 22) | ((padEn ? padAmt : 0) << 25); // data_size=2B
  g1[1] = (cols & 0xffff) << 16;                      // tensor_dim0[15:0]
  g1[2] = (rows & 0xffff) << 16;                      // tensor_dim1[15:0]
  g1[3] = (cols & 0xffff) << 16;                      // tile_dim0
  g1[4] = (rows & 0xffff);                            // tile_dim1 (tile_dim2=0)
  g1[5] = cols;                                       // tensor_dim0_stride[31:0]
  i32x4 z4 = {};
  i32x8 z8 = {};
  __builtin_amdgcn_tensor_load_to_lds(g0, g1, z4, z4, z8, 0);
}
#endif

// Stage a weight into LDS: TDM (wave 0 issues) or cooperative copy fallback.
__device__ __forceinline__ void stage_weight(const _Float16* src, _Float16* dst,
                                             int rows, int cols, int ldw, int tid, int w) {
#if HAVE_TDM
  (void)tid;
  if (w == 0) tdm_load_w(src, dst, rows, cols, ldw);
#else
  (void)w;
  load_w_lds(src, dst, rows, cols, ldw, tid);
#endif
}

__device__ __forceinline__ void stage_wait(int w) {
#if HAVE_TDM
  if (w == 0) __builtin_amdgcn_s_wait_tensorcnt(0);
#else
  (void)w;
#endif
}

__global__ __launch_bounds__(256)
void f32_to_f16_kernel(const float* __restrict__ src, _Float16* __restrict__ dst, int n) {
  int i = blockIdx.x * 256 + threadIdx.x;
  if (i < n) dst[i] = (_Float16)src[i];
}

// Pre-attention LayerNorm of node features -> f16 (one wave per node)
__global__ __launch_bounds__(256)
void node_ln_kernel(const float* __restrict__ nf, const float* __restrict__ g,
                    const float* __restrict__ b, _Float16* __restrict__ nh16, int n) {
  int wave = (blockIdx.x * 256 + threadIdx.x) >> 5;
  int lane = threadIdx.x & 31;
  if (wave >= n) return;
  const float* row = nf + wave * HH;
  float4 x = *(const float4*)(row + lane * 4);
  float s1 = x.x + x.y + x.z + x.w;
  float s2 = x.x * x.x + x.y * x.y + x.z * x.z + x.w * x.w;
  for (int m = 1; m <= 16; m <<= 1) { s1 += __shfl_xor(s1, m); s2 += __shfl_xor(s2, m); }
  float mean = s1 * (1.0f / 128.0f);
  float rstd = rsqrtf(s2 * (1.0f / 128.0f) - mean * mean + 1e-5f);
  float xv[4] = {x.x, x.y, x.z, x.w};
#pragma unroll
  for (int t = 0; t < 4; ++t) {
    int c = lane * 4 + t;
    nh16[wave * HH + c] = (_Float16)((xv[t] - mean) * rstd * g[c] + b[c]);
  }
}

// ---------------- fused per-node-group block (4 nodes = 128 rows per WG) ----------------
__global__ __launch_bounds__(256)
void edge_block_kernel(const float* __restrict__ ea, const float* __restrict__ ef,
                       const float* __restrict__ nf, const int* __restrict__ nbrs,
                       const int* __restrict__ numv,
                       const _Float16* __restrict__ nh16,
                       const _Float16* wE, const _Float16* wN, const _Float16* wM,
                       const _Float16* wQ, const _Float16* wO,
                       const _Float16* wE1, const _Float16* wE2,
                       const float* bE, const float* bN, const float* bM,
                       const float* bQ, const float* bO,
                       const float* bE1, const float* bE2,
                       const float* gFE, const float* bFE,
                       float* __restrict__ outN, float* __restrict__ outE) {
  extern __shared__ char smem[];
  _Float16* lAct = (_Float16*)smem;            // 128 x 264
  _Float16* lW   = lAct + 128 * 264;           // 50176 halves (weights)
  _Float16* lEH  = lW + 50176;                 // 128 x 136 (edge_hidden)
  _Float16* lQKV = lEH + 128 * 136;            // 128 x 392 (qkv / FFN hidden)
  float* nodeAcc = (float*)(lQKV + 128 * 392); // 4 x 128

  const int tid = threadIdx.x;
  const int w = tid >> 5, lane = tid & 31;
  const int nodeBase = blockIdx.x * 4;
  const int colL = lane & 15;
  const int rowbase = 16 * w + ((lane >> 4) << 3);
  const int nl = w >> 1;
  const int gnode = nodeBase + nl;

  // ---- P1: edge_attr -> f16 lAct[:, :128]; W_edge -> lW (TDM overlaps); zero nodeAcc ----
  stage_weight(wE, lW, 128, 128, 136, tid, w);
  {
    const float4* eaB = (const float4*)(ea + (size_t)nodeBase * KK * EE);
    for (int idx = tid; idx < 128 * 32; idx += 256) {
      int r = idx >> 5, c4 = idx & 31;
      float4 x = eaB[idx];
      _Float16* d = lAct + r * 264 + (c4 << 2);
      d[0] = (_Float16)x.x; d[1] = (_Float16)x.y; d[2] = (_Float16)x.z; d[3] = (_Float16)x.w;
    }
    for (int idx = tid; idx < 512; idx += 256) nodeAcc[idx] = 0.0f;
    if (lane == 0) __builtin_prefetch(wN + w * 512, 0, 1);
  }
  stage_wait(w);
  __syncthreads();

  // ---- P2: edge_hidden = gelu(edge_attr @ W_edge + b) -> lEH ----
  {
    AFr<4> af = load_afrags<4>(lAct + 16 * w * 264, 264, lane);
    for (int j = 0; j < 8; ++j) {
      v8f acc = wmma_btile<4>(af, lW + 16 * j, 136, lane);
      int col = 16 * j + colL;
      float bb = bE[col];
#pragma unroll
      for (int v = 0; v < 8; ++v)
        lEH[(rowbase + v) * 136 + col] = (_Float16)gelu_f(acc[v] + bb);
    }
  }
  __syncthreads();

  // ---- P3: lAct = [center | nbr] gather; W_node -> lW ----
  stage_weight(wN, lW, 256, 128, 136, tid, w);
  for (int idx = tid; idx < 128 * 32; idx += 256) {
    int r = idx >> 5, col = (idx & 31) << 3;
    int gn = nodeBase + (r >> 5);
    const _Float16* srcp;
    if (col < 128) srcp = nh16 + (size_t)gn * HH + col;
    else {
      int nb = nbrs[gn * KK + (r & 31)];
      srcp = nh16 + (size_t)nb * HH + (col - 128);
    }
    *(uint4*)(lAct + r * 264 + col) = *(const uint4*)srcp;
  }
  stage_wait(w);
  __syncthreads();

  {
    float tbuf[8][8];
    // ---- P4: node_hid = gelu(node_cat @ W_node + b) ----
    {
      AFr<8> af = load_afrags<8>(lAct + 16 * w * 264, 264, lane);
#pragma unroll
      for (int j = 0; j < 8; ++j) {
        v8f acc = wmma_btile<8>(af, lW + 16 * j, 136, lane);
        int col = 16 * j + colL;
        float bb = bN[col];
#pragma unroll
        for (int v = 0; v < 8; ++v) tbuf[j][v] = gelu_f(acc[v] + bb);
      }
    }
    __syncthreads();
    stage_weight(wM, lW, 256, 128, 136, tid, w);
#pragma unroll
    for (int j = 0; j < 8; ++j) {
      int col = 128 + 16 * j + colL;
#pragma unroll
      for (int v = 0; v < 8; ++v)
        lAct[(rowbase + v) * 264 + col] = (_Float16)tbuf[j][v];
    }
    for (int idx = tid; idx < 128 * 16; idx += 256) {
      int r = idx >> 4, col = (idx & 15) << 3;
      *(uint4*)(lAct + r * 264 + col) = *(const uint4*)(lEH + r * 136 + col);
    }
    stage_wait(w);
    __syncthreads();

    // ---- P5: msg = gelu([edge_hidden|node_hid] @ W_msg + b) ----
    {
      AFr<8> af = load_afrags<8>(lAct + 16 * w * 264, 264, lane);
#pragma unroll
      for (int j = 0; j < 8; ++j) {
        v8f acc = wmma_btile<8>(af, lW + 16 * j, 136, lane);
        int col = 16 * j + colL;
        float bb = bM[col];
#pragma unroll
        for (int v = 0; v < 8; ++v) tbuf[j][v] = gelu_f(acc[v] + bb);
      }
    }
    __syncthreads();
    stage_weight(wQ, lW, 128, 384, 384, tid, w);   // unpadded: ldw = 384
#pragma unroll
    for (int j = 0; j < 8; ++j) {
      int col = 16 * j + colL;
#pragma unroll
      for (int v = 0; v < 8; ++v)
        lAct[(rowbase + v) * 264 + col] = (_Float16)tbuf[j][v];
    }
    stage_wait(w);
    __syncthreads();
  }

  // ---- P6: qkv = msg @ W_qkv + b -> lQKV (f16) ----
  {
    AFr<4> af = load_afrags<4>(lAct + 16 * w * 264, 264, lane);
    for (int j = 0; j < 24; ++j) {
      v8f acc = wmma_btile<4>(af, lW + 16 * j, 384, lane);
      int col = 16 * j + colL;
      float bb = bQ[col];
#pragma unroll
      for (int v = 0; v < 8; ++v)
        lQKV[(rowbase + v) * 392 + col] = (_Float16)(acc[v] + bb);
    }
  }
  __syncthreads();

  // ---- P7: windowed MHA (per-lane query rows); W_out DMA overlaps ----
  stage_weight(wO, lW, 128, 128, 136, tid, w);
  for (int tt = tid; tt < 1024; tt += 256) {
    int q = tt & 31, ph = tt >> 5;
    int pnl = ph >> 3, h = ph & 7;
    int qrow = pnl * 32 + q;
    int nv = numv[nodeBase + pnl];
    const _Float16* qp = lQKV + qrow * 392 + h * 16;
    float qv[16];
#pragma unroll
    for (int d = 0; d < 16; ++d) qv[d] = (float)qp[d];
    float s[32]; float mx = -3.0e38f;
#pragma unroll
    for (int k = 0; k < 32; ++k) {
      const _Float16* kp = lQKV + (pnl * 32 + k) * 392 + 128 + h * 16;
      float d0 = 0.0f;
#pragma unroll
      for (int d = 0; d < 16; ++d) d0 += qv[d] * (float)kp[d];
      float sc = d0 * 0.25f;
      if (k >= nv) sc = -1.0e9f;
      s[k] = sc; mx = fmaxf(mx, sc);
    }
    float den = 0.0f;
#pragma unroll
    for (int k = 0; k < 32; ++k) { s[k] = __expf(s[k] - mx); den += s[k]; }
    float inv = __builtin_amdgcn_rcpf(den);
    float ctx[16];
#pragma unroll
    for (int d = 0; d < 16; ++d) ctx[d] = 0.0f;
#pragma unroll
    for (int k = 0; k < 32; ++k) {
      const _Float16* vp = lQKV + (pnl * 32 + k) * 392 + 256 + h * 16;
      float p = s[k];
#pragma unroll
      for (int d = 0; d < 16; ++d) ctx[d] += p * (float)vp[d];
    }
    _Float16* op = lAct + qrow * 264 + h * 16;
#pragma unroll
    for (int d = 0; d < 16; ++d) op[d] = (_Float16)(ctx[d] * inv);
  }
  stage_wait(w);
  __syncthreads();

  // ---- P8: edge_out = ctx @ W_out + b; residual, masked scatter, LN(fe) ----
  float res[8][8];
  {
    const float* efB = ef + (size_t)nodeBase * KK * HH;
    int nv = numv[gnode];
    AFr<4> af = load_afrags<4>(lAct + 16 * w * 264, 264, lane);
#pragma unroll
    for (int j = 0; j < 8; ++j) {
      v8f acc = wmma_btile<4>(af, lW + 16 * j, 136, lane);
      int col = 16 * j + colL;
      float bo = bO[col];
      float asum = 0.0f;
#pragma unroll
      for (int v = 0; v < 8; ++v) {
        int row = rowbase + v;
        float eo = acc[v] + bo;
        if ((row & 31) < nv) asum += eo;           // mask applies to scatter only
        res[j][v] = eo + efB[row * HH + col];      // residual (unmasked)
      }
      atomicAdd(&nodeAcc[nl * 128 + col], asum);
    }
    float mu[8], rs[8];
#pragma unroll
    for (int v = 0; v < 8; ++v) {
      float s1 = 0.0f, s2 = 0.0f;
#pragma unroll
      for (int j = 0; j < 8; ++j) { float x = res[j][v]; s1 += x; s2 += x * x; }
      s1 += __shfl_xor(s1, 1); s1 += __shfl_xor(s1, 2); s1 += __shfl_xor(s1, 4); s1 += __shfl_xor(s1, 8);
      s2 += __shfl_xor(s2, 1); s2 += __shfl_xor(s2, 2); s2 += __shfl_xor(s2, 4); s2 += __shfl_xor(s2, 8);
      float mean = s1 * (1.0f / 128.0f);
      mu[v] = mean;
      rs[v] = rsqrtf(s2 * (1.0f / 128.0f) - mean * mean + 1e-5f);
    }
#pragma unroll
    for (int j = 0; j < 8; ++j) {
      int col = 16 * j + colL;
      float g = gFE[col], bb = bFE[col];
#pragma unroll
      for (int v = 0; v < 8; ++v)
        lAct[(rowbase + v) * 264 + col] = (_Float16)((res[j][v] - mu[v]) * rs[v] * g + bb);
    }
  }
  __syncthreads();

  // ---- P9: node_res -> outN (We1 DMA overlaps) ----
  stage_weight(wE1, lW, 128, 256, 264, tid, w);
  for (int idx = tid; idx < 512; idx += 256) {
    int nn = idx >> 7, col = idx & 127;
    int g = nodeBase + nn;
    outN[g * HH + col] = nodeAcc[idx] + nf[g * HH + col];
  }
  stage_wait(w);
  __syncthreads();

  // ---- P10: edge FFN hidden = gelu(ln_e @ We1 + b) -> lQKV ----
  {
    AFr<4> af = load_afrags<4>(lAct + 16 * w * 264, 264, lane);
    for (int j = 0; j < 16; ++j) {
      v8f acc = wmma_btile<4>(af, lW + 16 * j, 264, lane);
      int col = 16 * j + colL;
      float bb = bE1[col];
#pragma unroll
      for (int v = 0; v < 8; ++v)
        lQKV[(rowbase + v) * 392 + col] = (_Float16)gelu_f(acc[v] + bb);
    }
  }
  __syncthreads();
  stage_weight(wE2, lW, 256, 128, 136, tid, w);
  stage_wait(w);
  __syncthreads();

  // ---- P11: edge final = hidden @ We2 + b + edge_res -> global ----
  {
    AFr<8> af = load_afrags<8>(lQKV + 16 * w * 392, 392, lane);
#pragma unroll
    for (int j = 0; j < 8; ++j) {
      v8f acc = wmma_btile<8>(af, lW + 16 * j, 136, lane);
      int col = 16 * j + colL;
      float bb = bE2[col];
#pragma unroll
      for (int v = 0; v < 8; ++v) {
        int row = rowbase + v;
        outE[((size_t)nodeBase * KK + row) * HH + col] = acc[v] + bb + res[j][v];
      }
    }
  }
}

// ---------------- node FFN: out += gelu(ln(out)@Wn1+b1)@Wn2+b2 ----------------
__global__ __launch_bounds__(256)
void node_ffn_kernel(float* __restrict__ outN, const float* __restrict__ gFN,
                     const float* __restrict__ bFN,
                     const _Float16* __restrict__ wN1, const _Float16* __restrict__ wN2,
                     const float* __restrict__ bn1, const float* __restrict__ bn2, int n) {
  extern __shared__ char smem[];
  _Float16* lAct = (_Float16*)smem;      // 128 x 136
  _Float16* lW   = lAct + 128 * 136;     // 34816
  _Float16* lHid = lW + 34816;           // 128 x 264
  const int tid = threadIdx.x, w = tid >> 5, lane = tid & 31;
  const int base = blockIdx.x * 128;
  stage_weight(wN1, lW, 128, 256, 264, tid, w);
  for (int r = w; r < 128; r += 8) {
    int g = base + r;
    float4 x = make_float4(0.f, 0.f, 0.f, 0.f);
    if (g < n) x = *(const float4*)(outN + g * HH + lane * 4);
    float s1 = x.x + x.y + x.z + x.w;
    float s2 = x.x * x.x + x.y * x.y + x.z * x.z + x.w * x.w;
    for (int m = 1; m <= 16; m <<= 1) { s1 += __shfl_xor(s1, m); s2 += __shfl_xor(s2, m); }
    float mean = s1 * (1.0f / 128.0f);
    float rstd = rsqrtf(s2 * (1.0f / 128.0f) - mean * mean + 1e-5f);
    float xv[4] = {x.x, x.y, x.z, x.w};
#pragma unroll
    for (int t = 0; t < 4; ++t) {
      int c = lane * 4 + t;
      lAct[r * 136 + c] = (_Float16)((xv[t] - mean) * rstd * gFN[c] + bFN[c]);
    }
  }
  stage_wait(w);
  __syncthreads();
  const int colL = lane & 15;
  const int rowbase = 16 * w + ((lane >> 4) << 3);
  {
    AFr<4> af = load_afrags<4>(lAct + 16 * w * 136, 136, lane);
    for (int j = 0; j < 16; ++j) {
      v8f acc = wmma_btile<4>(af, lW + 16 * j, 264, lane);
      int col = 16 * j + colL;
      float b1 = bn1[col];
#pragma unroll
      for (int v = 0; v < 8; ++v)
        lHid[(rowbase + v) * 264 + col] = (_Float16)gelu_f(acc[v] + b1);
    }
  }
  __syncthreads();
  stage_weight(wN2, lW, 256, 128, 136, tid, w);
  stage_wait(w);
  __syncthreads();
  {
    AFr<8> af = load_afrags<8>(lHid + 16 * w * 264, 264, lane);
#pragma unroll
    for (int j = 0; j < 8; ++j) {
      v8f acc = wmma_btile<8>(af, lW + 16 * j, 136, lane);
      int col = 16 * j + colL;
      float b2 = bn2[col];
#pragma unroll
      for (int v = 0; v < 8; ++v) {
        int g = base + rowbase + v;
        if (g < n) outN[g * HH + col] += acc[v] + b2;
      }
    }
  }
}

extern "C" void kernel_launch(void* const* d_in, const int* in_sizes, int n_in,
                              void* d_out, int out_size, void* d_ws, size_t ws_size,
                              hipStream_t stream) {
  const float* nf = (const float*)d_in[0];
  const float* ef = (const float*)d_in[1];
  const float* ea = (const float*)d_in[2];
  const int* nbrs = (const int*)d_in[3];
  const int* numv = (const int*)d_in[4];

  _Float16* wsH  = (_Float16*)d_ws;
  _Float16* nh16 = wsH;
  _Float16* wE   = wsH + (size_t)NN * HH;
  _Float16* wN   = wE  + 128 * 128;
  _Float16* wM   = wN  + 256 * 128;
  _Float16* wQ   = wM  + 256 * 128;
  _Float16* wO   = wQ  + 128 * 384;
  _Float16* wN1  = wO  + 128 * 128;
  _Float16* wN2  = wN1 + 128 * 256;
  _Float16* wE1  = wN2 + 256 * 128;
  _Float16* wE2  = wE1 + 128 * 256;

  auto cvt = [&](int idx, _Float16* dst, int n) {
    f32_to_f16_kernel<<<(n + 255) / 256, 256, 0, stream>>>((const float*)d_in[idx], dst, n);
  };
  cvt(5,  wE,  128 * 128);
  cvt(7,  wN,  256 * 128);
  cvt(9,  wM,  256 * 128);
  cvt(11, wQ,  128 * 384);
  cvt(13, wO,  128 * 128);
  cvt(21, wN1, 128 * 256);
  cvt(23, wN2, 256 * 128);
  cvt(25, wE1, 128 * 256);
  cvt(27, wE2, 256 * 128);

  node_ln_kernel<<<NN / 8, 256, 0, stream>>>(nf, (const float*)d_in[15],
                                             (const float*)d_in[16], nh16, NN);

  float* outN = (float*)d_out;
  float* outE = outN + (size_t)NN * HH;

  const size_t SMEM_C = (size_t)(128 * 264 + 50176 + 128 * 136 + 128 * 392) * sizeof(_Float16)
                      + 512 * sizeof(float);   // 305,152 B
  const size_t SMEM_D = (size_t)(128 * 136 + 34816 + 128 * 264) * sizeof(_Float16); // 172,032 B
  (void)hipFuncSetAttribute((const void*)edge_block_kernel,
                            hipFuncAttributeMaxDynamicSharedMemorySize, (int)SMEM_C);
  (void)hipFuncSetAttribute((const void*)node_ffn_kernel,
                            hipFuncAttributeMaxDynamicSharedMemorySize, (int)SMEM_D);

  edge_block_kernel<<<NN / 4, 256, SMEM_C, stream>>>(
      ea, ef, nf, nbrs, numv, nh16,
      wE, wN, wM, wQ, wO, wE1, wE2,
      (const float*)d_in[6],  (const float*)d_in[8],  (const float*)d_in[10],
      (const float*)d_in[12], (const float*)d_in[14],
      (const float*)d_in[26], (const float*)d_in[28],
      (const float*)d_in[19], (const float*)d_in[20],
      outN, outE);

  node_ffn_kernel<<<(NN + 127) / 128, 256, SMEM_D, stream>>>(
      outN, (const float*)d_in[17], (const float*)d_in[18],
      wN1, wN2, (const float*)d_in[22], (const float*)d_in[24], NN);
}
